// AsymmetricGraphAttentionLayer_31628139167942
// MI455X (gfx1250) — matrix operand order repaired and checked by reading, not compiled, not run on hardware
//
#include <hip/hip_runtime.h>

#define B_  4
#define N_  4096
#define F_  128
#define H_  4
#define HD_ 32

typedef __attribute__((ext_vector_type(16))) __bf16       v16bf;
typedef __attribute__((ext_vector_type(8)))  __bf16       v8bf;
typedef __attribute__((ext_vector_type(2)))  __bf16       v2bf;
typedef __attribute__((ext_vector_type(8)))  float        v8f;
typedef __attribute__((ext_vector_type(4)))  float        v4f;
typedef __attribute__((ext_vector_type(4)))  unsigned int v4u;

#define WSP_STRIDE 20   // dword stride per column (16 used + 4 pad)

// ---------------------------------------------------------------------------
// Kernel 1: Wx = x @ W   (16384 x 128 x 128), output bf16
// W staged in LDS as bf16 k-pairs in B-fragment order (converted once/block).
// ---------------------------------------------------------------------------
__global__ __launch_bounds__(256)
void gat_gemm_xw(const float* __restrict__ x, const float* __restrict__ W,
                 __bf16* __restrict__ WxBF) {
    __shared__ v2bf WsP[128 * WSP_STRIDE];
    const int tid    = threadIdx.x;
    const int wave   = tid >> 5;
    const int lane   = tid & 31;
    const int lanelo = lane & 15;
    const int hi     = (lane >> 4) & 1;
    const int rowBase = blockIdx.x * 128 + wave * 16;
    const int kb1 = hi ? 8 : 0;
    const int cw  = tid & 127;        // column this thread packs
    const int tq  = tid >> 7;         // 0..1

    v8f acc[8] = {};
    const float* xrow = x + (size_t)(rowBase + lanelo) * F_;

    for (int kt = 0; kt < 4; ++kt) {
        const int k0 = kt * 32;
        __syncthreads();
        #pragma unroll
        for (int s = 0; s < 8; ++s) {
            const int t = tq * 8 + s;                 // k-pair index 0..15
            const float w0 = W[(k0 + 2 * t) * F_ + cw];
            const float w1 = W[(k0 + 2 * t + 1) * F_ + cw];
            v2bf p; p[0] = (__bf16)w0; p[1] = (__bf16)w1;
            WsP[cw * WSP_STRIDE + t] = p;
        }
        __syncthreads();

        // A fragment: 16x32 bf16; lanes l / l+16 jointly hold row l%16
        const float4 a0 = *reinterpret_cast<const float4*>(xrow + k0 + kb1);
        const float4 a1 = *reinterpret_cast<const float4*>(xrow + k0 + kb1 + 4);
        const float4 a2 = *reinterpret_cast<const float4*>(xrow + k0 + kb1 + 16);
        const float4 a3 = *reinterpret_cast<const float4*>(xrow + k0 + kb1 + 20);
        float av[16];
        *reinterpret_cast<float4*>(av + 0)  = a0;
        *reinterpret_cast<float4*>(av + 4)  = a1;
        *reinterpret_cast<float4*>(av + 8)  = a2;
        *reinterpret_cast<float4*>(av + 12) = a3;
        v16bf af;
        #pragma unroll
        for (int e = 0; e < 16; ++e) af[e] = (__bf16)av[e];

        #pragma unroll
        for (int ct = 0; ct < 8; ++ct) {
            const int c = ct * 16 + lanelo;
            const v4u* col = reinterpret_cast<const v4u*>(&WsP[c * WSP_STRIDE + hi * 4]);
            const v8bf q0 = __builtin_bit_cast(v8bf, col[0]);
            const v8bf q1 = __builtin_bit_cast(v8bf, col[2]);
            v16bf bf;
            #pragma unroll
            for (int e = 0; e < 8; ++e) { bf[e] = q0[e]; bf[8 + e] = q1[e]; }
            acc[ct] = __builtin_amdgcn_wmma_f32_16x16x32_bf16(
                false, af, false, bf, (short)0, acc[ct], false, false);
        }
    }

    #pragma unroll
    for (int ct = 0; ct < 8; ++ct) {
        const int c = ct * 16 + lanelo;
        #pragma unroll
        for (int r = 0; r < 8; ++r) {
            const int grow = rowBase + r + hi * 8;
            WxBF[(size_t)grow * F_ + c] = (__bf16)acc[ct][r];
        }
    }
}

// ---------------------------------------------------------------------------
// Kernel 2: s_l[b,h,n] = Wx[b,n,h*32:] . a_l[h] ; s_r likewise.
// Also writes the d-major transpose WxT[(b*H+h)*HD + d][n] used by the
// attention kernel's WMMA B-fragments (contiguous-K loads).
// ---------------------------------------------------------------------------
__global__ __launch_bounds__(256)
void gat_scores(const __bf16* __restrict__ WxBF, const float* __restrict__ a,
                float* __restrict__ sl, float* __restrict__ sr,
                __bf16* __restrict__ WxT) {
    const int tid = blockIdx.x * 256 + threadIdx.x;   // < B*H*N
    const int n  = tid & (N_ - 1);
    const int bh = tid >> 12;                         // N_ = 4096
    const int h  = bh & (H_ - 1);
    const int b  = bh >> 2;
    const __bf16* wrow = WxBF + ((size_t)(b * N_ + n)) * F_ + h * HD_;
    float suml = 0.f, sumr = 0.f;
    #pragma unroll
    for (int d = 0; d < HD_; ++d) {
        const __bf16 wb = wrow[d];
        const float w = (float)wb;
        suml += w * a[h * 2 * HD_ + d];
        sumr += w * a[h * 2 * HD_ + HD_ + d];
        WxT[((size_t)bh * HD_ + d) * N_ + n] = wb;    // coalesced across n
    }
    sl[tid] = suml;
    sr[tid] = sumr;
}

// ---------------------------------------------------------------------------
// Kernel 3: flash-style masked softmax attention + aggregation h = attn @ Wx
// Block: 256 threads = 8 waves; block handles (b, h, 128 rows).
// Each wave owns 16 rows, streams j in tiles of 64 (two WMMA K-slabs per
// softmax-stats update -> 4 WMMAs per iteration). adj streamed non-temporal.
// Row stats live replicated in lanes l and l+16 (A-fragment row pairing).
// ---------------------------------------------------------------------------
__global__ __launch_bounds__(256)
void gat_attn(const float* __restrict__ adj, const __bf16* __restrict__ WxT,
              const float* __restrict__ sl, const float* __restrict__ sr,
              __bf16* __restrict__ Hbf) {
    __shared__ float4 srs4[N_ / 4];
    float* srs = reinterpret_cast<float*>(srs4);

    const int tid    = threadIdx.x;
    const int wave   = tid >> 5;
    const int lane   = tid & 31;
    const int lanelo = lane & 15;
    const int hi     = (lane >> 4) & 1;

    const int rt = blockIdx.x & 31;           // row tile of 128
    const int h  = (blockIdx.x >> 5) & (H_ - 1);
    const int b  = blockIdx.x >> 7;           // / (32*H_)
    const size_t bh = (size_t)(b * H_ + h);

    // stage s_r[b,h,:] into LDS
    const float* srg = sr + bh * N_;
    #pragma unroll
    for (int s = 0; s < 16; ++s) {
        const int idx = tid + 256 * s;
        srs[idx] = srg[idx];
    }
    __syncthreads();

    const int rowBase = rt * 128 + wave * 16;
    const int i = rowBase + lanelo;
    const float sli = sl[bh * N_ + i];
    const float* adjrow = adj + ((size_t)b * N_ + i) * N_;
    const int kb1 = hi ? 16 : 0;

    // d-major Wx rows for the two 16-col D tiles this wave produces
    const __bf16* wtr0 = WxT + (bh * HD_ + lanelo) * N_;        // d = lanelo
    const __bf16* wtr1 = WxT + (bh * HD_ + 16 + lanelo) * N_;   // d = lanelo+16

    float m = -1e30f, l = 0.f;
    v8f acc0 = {}, acc1 = {};

    for (int j0 = 0; j0 < N_; j0 += 64) {
        if (j0 + 64 < N_)
            __builtin_prefetch(adjrow + j0 + 64, 0, 0);

        float aj[32], sv[32];
        #pragma unroll
        for (int s = 0; s < 2; ++s) {           // two 32-wide K-slabs
            const int base = j0 + 32 * s + kb1;
            *reinterpret_cast<v4f*>(aj + 16 * s + 0) =
                __builtin_nontemporal_load(reinterpret_cast<const v4f*>(adjrow + base));
            *reinterpret_cast<v4f*>(aj + 16 * s + 4) =
                __builtin_nontemporal_load(reinterpret_cast<const v4f*>(adjrow + base + 4));
            *reinterpret_cast<v4f*>(aj + 16 * s + 8) =
                __builtin_nontemporal_load(reinterpret_cast<const v4f*>(adjrow + base + 8));
            *reinterpret_cast<v4f*>(aj + 16 * s + 12) =
                __builtin_nontemporal_load(reinterpret_cast<const v4f*>(adjrow + base + 12));
            *reinterpret_cast<float4*>(sv + 16 * s + 0) =
                *reinterpret_cast<const float4*>(srs + base);
            *reinterpret_cast<float4*>(sv + 16 * s + 4) =
                *reinterpret_cast<const float4*>(srs + base + 4);
            *reinterpret_cast<float4*>(sv + 16 * s + 8) =
                *reinterpret_cast<const float4*>(srs + base + 8);
            *reinterpret_cast<float4*>(sv + 16 * s + 12) =
                *reinterpret_cast<const float4*>(srs + base + 12);
        }

        float ev[32];
        float vmax = -1e30f;
        #pragma unroll
        for (int e = 0; e < 32; ++e) {
            float t = sli + sv[e];
            t = t > 0.f ? t : 0.2f * t;                 // leaky_relu(0.2)
            ev[e] = (aj[e] != 0.f) ? t : -1e30f;        // mask
            vmax = fmaxf(vmax, ev[e]);
        }
        vmax = fmaxf(vmax, __shfl_xor(vmax, 16, 32));   // combine row halves

        // online-softmax rescale; skipped (wave-uniform) once maxima settle
        float mnew, scale;
        if (__ballot(vmax > m)) {
            mnew  = fmaxf(m, vmax);
            scale = __expf(m - mnew);
            #pragma unroll
            for (int r = 0; r < 8; ++r) {
                const float sc = __shfl(scale, r + hi * 8, 32);
                acc0[r] *= sc;
                acc1[r] *= sc;
            }
        } else {
            mnew  = m;
            scale = 1.0f;
        }

        float rs = 0.f;
        v16bf af0, af1;
        #pragma unroll
        for (int e = 0; e < 16; ++e) {
            const float p0 = (aj[e] != 0.f)      ? __expf(ev[e] - mnew)      : 0.f;
            const float p1 = (aj[16 + e] != 0.f) ? __expf(ev[16 + e] - mnew) : 0.f;
            rs += p0 + p1;
            af0[e] = (__bf16)p0;
            af1[e] = (__bf16)p1;
        }
        rs += __shfl_xor(rs, 16, 32);
        l = l * scale + rs;
        m = mnew;

        // B fragments: 16 consecutive K=j values per lane, straight from WxT
        const int jb0 = j0 + hi * 16;
        const v16bf bf00 = *reinterpret_cast<const v16bf*>(wtr0 + jb0);
        const v16bf bf01 = *reinterpret_cast<const v16bf*>(wtr1 + jb0);
        const v16bf bf10 = *reinterpret_cast<const v16bf*>(wtr0 + jb0 + 32);
        const v16bf bf11 = *reinterpret_cast<const v16bf*>(wtr1 + jb0 + 32);

        acc0 = __builtin_amdgcn_wmma_f32_16x16x32_bf16(
            false, af0, false, bf00, (short)0, acc0, false, false);
        acc1 = __builtin_amdgcn_wmma_f32_16x16x32_bf16(
            false, af0, false, bf01, (short)0, acc1, false, false);
        acc0 = __builtin_amdgcn_wmma_f32_16x16x32_bf16(
            false, af1, false, bf10, (short)0, acc0, false, false);
        acc1 = __builtin_amdgcn_wmma_f32_16x16x32_bf16(
            false, af1, false, bf11, (short)0, acc1, false, false);
    }

    const float inv = 1.0f / l;   // self-loop guarantees l > 0
    __bf16* hrow = Hbf + (size_t)b * N_ * F_ + h * HD_;
    #pragma unroll
    for (int r = 0; r < 8; ++r) {
        const float iv = __shfl(inv, r + hi * 8, 32);
        const int grow = rowBase + r + hi * 8;
        hrow[(size_t)grow * F_ + lanelo]      = (__bf16)(acc0[r] * iv);
        hrow[(size_t)grow * F_ + lanelo + 16] = (__bf16)(acc1[r] * iv);
    }
}

// ---------------------------------------------------------------------------
// Kernel 4: out = elu(h @ Wo + bo)   (16384 x 128 x 128), f32 output
// ---------------------------------------------------------------------------
__global__ __launch_bounds__(256)
void gat_gemm_out(const __bf16* __restrict__ Hbf, const float* __restrict__ Wo,
                  const float* __restrict__ bo, float* __restrict__ out) {
    __shared__ v2bf WsP[128 * WSP_STRIDE];
    const int tid    = threadIdx.x;
    const int wave   = tid >> 5;
    const int lane   = tid & 31;
    const int lanelo = lane & 15;
    const int hi     = (lane >> 4) & 1;
    const int rowBase = blockIdx.x * 128 + wave * 16;
    const int kb1 = hi ? 8 : 0;
    const int cw  = tid & 127;
    const int tq  = tid >> 7;

    v8f acc[8] = {};
    const __bf16* hrow = Hbf + (size_t)(rowBase + lanelo) * F_;

    for (int kt = 0; kt < 4; ++kt) {
        const int k0 = kt * 32;
        __syncthreads();
        #pragma unroll
        for (int s = 0; s < 8; ++s) {
            const int t = tq * 8 + s;
            const float w0 = Wo[(k0 + 2 * t) * F_ + cw];
            const float w1 = Wo[(k0 + 2 * t + 1) * F_ + cw];
            v2bf p; p[0] = (__bf16)w0; p[1] = (__bf16)w1;
            WsP[cw * WSP_STRIDE + t] = p;
        }
        __syncthreads();

        // A fragment: two 16-byte loads of 8 consecutive bf16 each
        const v8bf a0 = __builtin_bit_cast(
            v8bf, *reinterpret_cast<const v4u*>(hrow + k0 + kb1));
        const v8bf a1 = __builtin_bit_cast(
            v8bf, *reinterpret_cast<const v4u*>(hrow + k0 + kb1 + 16));
        v16bf af;
        #pragma unroll
        for (int e = 0; e < 8; ++e) { af[e] = a0[e]; af[8 + e] = a1[e]; }

        #pragma unroll
        for (int ct = 0; ct < 8; ++ct) {
            const int c = ct * 16 + lanelo;
            const v4u* col = reinterpret_cast<const v4u*>(&WsP[c * WSP_STRIDE + hi * 4]);
            const v8bf q0 = __builtin_bit_cast(v8bf, col[0]);
            const v8bf q1 = __builtin_bit_cast(v8bf, col[2]);
            v16bf bf;
            #pragma unroll
            for (int e = 0; e < 8; ++e) { bf[e] = q0[e]; bf[8 + e] = q1[e]; }
            acc[ct] = __builtin_amdgcn_wmma_f32_16x16x32_bf16(
                false, af, false, bf, (short)0, acc[ct], false, false);
        }
    }

    #pragma unroll
    for (int ct = 0; ct < 8; ++ct) {
        const int c = ct * 16 + lanelo;
        const float bc = bo[c];
        #pragma unroll
        for (int r = 0; r < 8; ++r) {
            const int grow = rowBase + r + hi * 8;
            float v = acc[ct][r] + bc;
            v = v > 0.f ? v : (__expf(v) - 1.f);   // elu
            __builtin_nontemporal_store(v, &out[(size_t)grow * F_ + c]);
        }
    }
}

// ---------------------------------------------------------------------------
extern "C" void kernel_launch(void* const* d_in, const int* in_sizes, int n_in,
                              void* d_out, int out_size, void* d_ws, size_t ws_size,
                              hipStream_t stream) {
    const float* x   = (const float*)d_in[0];
    const float* adj = (const float*)d_in[1];
    const float* W   = (const float*)d_in[2];
    const float* a   = (const float*)d_in[3];
    const float* Wo  = (const float*)d_in[4];
    const float* bo  = (const float*)d_in[5];
    float* out = (float*)d_out;

    char* ws = (char*)d_ws;
    __bf16* WxBF = (__bf16*)ws;                                        // 4 MB
    __bf16* Hbf  = (__bf16*)(ws + (size_t)B_ * N_ * F_ * 2);           // 4 MB
    float*  sl   = (float*)(ws + (size_t)B_ * N_ * F_ * 4);            // 256 KB
    float*  sr   = sl + (size_t)B_ * H_ * N_;                          // 256 KB
    __bf16* WxT  = (__bf16*)(ws + (size_t)B_ * N_ * F_ * 4 +
                             (size_t)2 * B_ * H_ * N_ * 4);            // 4 MB

    gat_gemm_xw<<<(B_ * N_) / 128, 256, 0, stream>>>(x, W, WxBF);
    gat_scores<<<(B_ * H_ * N_) / 256, 256, 0, stream>>>(WxBF, a, sl, sr, WxT);
    gat_attn<<<B_ * H_ * (N_ / 128), 256, 0, stream>>>(adj, WxT, sl, sr, Hbf);
    gat_gemm_out<<<(B_ * N_) / 128, 256, 0, stream>>>(Hbf, Wo, bo, out);
}
